// SelfAttention_3289944949478
// MI455X (gfx1250) — matrix operand order repaired and checked by reading, not compiled, not run on hardware
//
#include <hip/hip_runtime.h>
#include <hip/hip_bf16.h>

#define BB 2
#define LL 2048
#define DDIM 1024
#define HH 16
#define HD 64
#define MM (BB*LL)   // 4096 rows

typedef __attribute__((ext_vector_type(16))) __bf16 v16bf;
typedef __attribute__((ext_vector_type(8)))  __bf16 v8bf;
typedef __attribute__((ext_vector_type(8)))  float  v8f;

// async-to-LDS builtin wants v4i pointers: AS1 (global) and AS3 (LDS)
typedef int v4i __attribute__((vector_size(16)));
typedef __attribute__((address_space(1))) v4i gv4i;
typedef __attribute__((address_space(3))) v4i lv4i;
#define GPTR(p) ((gv4i*)(p))
#define LPTR(p) ((lv4i*)(p))

#if defined(__gfx1250__) && __has_builtin(__builtin_amdgcn_global_load_async_to_lds_b128)
#define HAVE_ASYNC 1
#else
#define HAVE_ASYNC 0
#endif

__device__ inline v8f vzero(){ v8f z = {0.f,0.f,0.f,0.f,0.f,0.f,0.f,0.f}; return z; }
__device__ inline __bf16 bfc(float f){ return (__bf16)f; }

// load 16 bf16 as two 8-element (16B) chunks from p0 and p1
__device__ inline v16bf ld8x2(const __bf16* p0, const __bf16* p1){
  v8bf a = *(const v8bf*)p0;
  v8bf b = *(const v8bf*)p1;
  v16bf r;
  #pragma unroll
  for(int i=0;i<8;i++){ r[i]=a[i]; r[8+i]=b[i]; }
  return r;
}

// ---------------- weight packing: de-interleave + transpose + bf16 ----------
// Wt[n*K + k] = bf16( W[k*(N*inter) + n*inter + phase] )
__global__ void k_packw(const float* __restrict__ W, __bf16* __restrict__ Wt,
                        int N, int Kd, int inter, int phase){
  int total = N*Kd;
  for(int idx = blockIdx.x*blockDim.x + threadIdx.x; idx < total;
      idx += gridDim.x*blockDim.x){
    int n = idx / Kd, k = idx - n*Kd;
    Wt[idx] = bfc(W[(size_t)k*(size_t)(N*inter) + (size_t)n*inter + phase]);
  }
}

__global__ void k_cvt_bf16(const float* __restrict__ src, __bf16* __restrict__ dst, int n){
  for(int i = blockIdx.x*blockDim.x + threadIdx.x; i < n; i += gridDim.x*blockDim.x)
    dst[i] = bfc(src[i]);
}

// ---------------- RMS statistics per row -----------------------------------
__global__ __launch_bounds__(256)
void k_rms(const float* __restrict__ x, float* __restrict__ invr){
  int row = blockIdx.x;
  const float* p = x + (size_t)row*DDIM;
  float s = 0.f;
  for(int i = threadIdx.x; i < DDIM; i += 256){ float v = p[i]; s += v*v; }
  #pragma unroll
  for(int mk=1; mk<32; mk<<=1) s += __shfl_xor(s, mk, 32);
  __shared__ float red[8];
  int lane = threadIdx.x & 31, wv = threadIdx.x >> 5;
  if(lane == 0) red[wv] = s;
  __syncthreads();
  if(threadIdx.x == 0){
    float t = 0.f;
    #pragma unroll
    for(int i=0;i<8;i++) t += red[i];
    invr[row] = rsqrtf(t*(1.0f/DDIM) + 1e-6f);
  }
}

// ---------------- modulate: xm = rmsnorm(x)*(1+scale)+shift -> bf16 ---------
__global__ void k_mod(const float* __restrict__ x, const float* __restrict__ invr,
                      const float* __restrict__ g, const float* __restrict__ sc,
                      const float* __restrict__ sh, __bf16* __restrict__ xmb, int n){
  for(int i = blockIdx.x*blockDim.x + threadIdx.x; i < n; i += gridDim.x*blockDim.x){
    int row = i >> 10, d = i & 1023;
    float xn = x[i]*invr[row]*g[d];
    xmb[i] = bfc(xn*(1.f + sc[i]) + sh[i]);
  }
}

// ---------------- bf16 WMMA GEMM: C[M,N] = A[M,K] * Bt[N,K]^T ---------------
enum { EPI_F32 = 0, EPI_BF16 = 1, EPI_FINAL = 2 };

template<int EPI>
__global__ __launch_bounds__(256)
void k_gemm(const __bf16* __restrict__ A, const __bf16* __restrict__ Bt,
            float* __restrict__ Cf, __bf16* __restrict__ Cb,
            const float* __restrict__ resid, const float* __restrict__ gate,
            int Mdim, int Ndim, int Kdim){
  // stride 56 bf16 (112B): 16B-aligned rows, conflict-free fragment reads
  __shared__ __align__(16) __bf16 As[128][56];
  __shared__ __align__(16) __bf16 Bs[64][56];
  const int tid  = threadIdx.x;
  const int lane = tid & 31, wv = tid >> 5;
  const int wm = wv >> 1, wn = wv & 1;        // 4x2 wave grid over 128x64 tile
  const int r = lane & 15, hb = lane >> 4;
  const int kb = hb*8;
  const int m0 = blockIdx.y * 128, n0 = blockIdx.x * 64;

  v8f acc[2][2];
  #pragma unroll
  for(int i=0;i<2;i++){
    #pragma unroll
    for(int j=0;j<2;j++) acc[i][j] = vzero();
  }

  for(int k0 = 0; k0 < Kdim; k0 += 32){
    __syncthreads();
#if HAVE_ASYNC
    // CDNA5 async LDS fill: no VGPR staging, tracked by ASYNCcnt
    #pragma unroll
    for(int t=0;t<2;t++){
      int idx = tid + t*256;
      int ar = idx >> 2, ac = (idx & 3) * 8;
      __builtin_amdgcn_global_load_async_to_lds_b128(
          GPTR(&A[(size_t)(m0+ar)*Kdim + k0 + ac]), LPTR(&As[ar][ac]), 0, 0);
    }
    {
      int br = tid >> 2, bc = (tid & 3) * 8;
      __builtin_amdgcn_global_load_async_to_lds_b128(
          GPTR(&Bt[(size_t)(n0+br)*Kdim + k0 + bc]), LPTR(&Bs[br][bc]), 0, 0);
    }
    asm volatile("s_wait_asynccnt 0" ::: "memory");
#else
    #pragma unroll
    for(int t=0;t<2;t++){
      int idx = tid + t*256;
      int ar = idx >> 2, ac = (idx & 3) * 8;
      *(uint4*)&As[ar][ac] = *(const uint4*)&A[(size_t)(m0+ar)*Kdim + k0 + ac];
    }
    {
      int br = tid >> 2, bc = (tid & 3) * 8;
      *(uint4*)&Bs[br][bc] = *(const uint4*)&Bt[(size_t)(n0+br)*Kdim + k0 + bc];
    }
#endif
    __syncthreads();

    v16bf af[2], bfr[2];
    #pragma unroll
    for(int s=0;s<2;s++){
      int row = wm*32 + s*16 + r;
      af[s]  = ld8x2(&As[row][kb], &As[row][kb+16]);          // A frag layout
      int col = wn*32 + s*16 + r;
      bfr[s] = ld8x2(&Bs[col][kb], &Bs[col][kb+16]);          // B frag layout
    }
    #pragma unroll
    for(int i=0;i<2;i++){
      #pragma unroll
      for(int j=0;j<2;j++)
        acc[i][j] = __builtin_amdgcn_wmma_f32_16x16x32_bf16(
            false, af[i], false, bfr[j], (short)0, acc[i][j], false, false);
    }
  }

  #pragma unroll
  for(int i=0;i<2;i++){
    #pragma unroll
    for(int j=0;j<2;j++){
      #pragma unroll
      for(int v=0;v<8;v++){
        int gm = m0 + wm*32 + i*16 + v + hb*8;   // C frag: row = vgpr + 8*half
        int gn = n0 + wn*32 + j*16 + r;          // col = lane&15
        size_t o = (size_t)gm*Ndim + gn;
        float val = acc[i][j][v];
        if(EPI == EPI_F32)       Cf[o] = val;
        else if(EPI == EPI_BF16) Cb[o] = bfc(val);
        else                     Cf[o] = val + resid[o] + gate[o];
      }
    }
  }
}

// ---------------- RoPE + head regroup ---------------------------------------
// q[b,l,h,d] = q_flat[d*H+h]; rope with sin=pos[2d], cos=pos[2d+1]
// Q/K -> [b,h,l,d] ; V -> transposed [b,h,d,l] for contiguous PV A-fragments
__global__ __launch_bounds__(256)
void k_rope(const __bf16* __restrict__ qw, const __bf16* __restrict__ kw,
            const __bf16* __restrict__ vw, const float* __restrict__ pos,
            __bf16* __restrict__ Q, __bf16* __restrict__ Kh, __bf16* __restrict__ Vt){
  int row = blockIdx.x;                 // b*L + l
  int b = row >> 11, l = row & (LL-1);
  const size_t rb = (size_t)row * DDIM;
  #pragma unroll
  for(int it=0; it<4; it++){
    int idx = it*256 + threadIdx.x;     // 0..1023
    int h = idx >> 6, d = idx & 63;
    int f  = d*HH + h;
    int d2 = (d < 32) ? d + 32 : d - 32;
    float sgn = (d < 32) ? -1.f : 1.f;  // rotate_half: [-t_hi, t_lo]
    int f2 = d2*HH + h;
    float sv = pos[(size_t)row*128 + 2*d];
    float cv = pos[(size_t)row*128 + 2*d + 1];
    float qv = (float)qw[rb + f], qp = (float)qw[rb + f2];
    float kv = (float)kw[rb + f], kp = (float)kw[rb + f2];
    size_t oq = ((size_t)(b*HH + h)*LL + l)*HD + d;
    Q[oq]  = bfc(qv*cv + sgn*qp*sv);
    Kh[oq] = bfc(kv*cv + sgn*kp*sv);
    Vt[((size_t)(b*HH + h)*HD + d)*LL + l] = vw[rb + f];
  }
}

// ---------------- flash attention (bf16 WMMA, transposed online softmax) ----
// S^T = K(A) x Q^T(B): keys in C-frag rows, queries across lanes ->
// per-query softmax stats are per-lane scalars (VALU tree + one shfl_xor(16)).
// O^T = V^T(A) x P^T(B): alpha/1/l rescale is elementwise per lane.
__global__ __launch_bounds__(128)
void k_attn(const __bf16* __restrict__ Q, const __bf16* __restrict__ Kh,
            const __bf16* __restrict__ Vt, __bf16* __restrict__ O){
  __shared__ __align__(16) __bf16 Pb[4][16][56];   // per-wave P^T [query][key]
  const int lane = threadIdx.x & 31, wv = threadIdx.x >> 5;
  const int r = lane & 15, hb = lane >> 4, kb = hb*8;
  const int bh = blockIdx.y;            // b*H + h
  const int b = bh >> 4, h = bh & 15;
  const int q0 = blockIdx.x*64 + wv*16; // 16 queries per wave
  const __bf16* Qp = Q  + (size_t)bh*LL*HD;
  const __bf16* Kp = Kh + (size_t)bh*LL*HD;
  const __bf16* Vp = Vt + (size_t)bh*HD*LL;

  // Q as B-fragments (d x query): lane = query r, d chunks kb and kb+16 per half
  v16bf bq[2];
  #pragma unroll
  for(int c=0;c<2;c++){
    const __bf16* p = Qp + (size_t)(q0 + r)*HD + c*32 + kb;
    bq[c] = ld8x2(p, p + 16);
  }

  v8f o[4];                       // O^T frags: rows = dim chunk, lanes = query
  #pragma unroll
  for(int c=0;c<4;c++) o[c] = vzero();
  float mx = -1e30f, lsum = 0.f;  // per-lane stats for query q0+r

  for(int j0 = 0; j0 < LL; j0 += 32){
    // S^T[key, query]: two 16-key subtiles
    v8f s[2]; s[0] = vzero(); s[1] = vzero();
    #pragma unroll
    for(int st=0; st<2; st++){
      #pragma unroll
      for(int c=0; c<2; c++){
        const __bf16* p = Kp + (size_t)(j0 + st*16 + r)*HD + c*32 + kb;
        v16bf ak = ld8x2(p, p + 16);   // A frag: lane = key row
        s[st] = __builtin_amdgcn_wmma_f32_16x16x32_bf16(
            false, ak, false, bq[c], (short)0, s[st], false, false);
      }
    }
    // per-lane max over this lane's 16 keys (pure VALU), merge halves once
    float tmax = -1e30f;
    #pragma unroll
    for(int st=0; st<2; st++){
      #pragma unroll
      for(int v=0; v<8; v++){
        s[st][v] *= 0.125f;            // 1/sqrt(64)
        tmax = fmaxf(tmax, s[st][v]);
      }
    }
    tmax = fmaxf(tmax, __shfl_xor(tmax, 16, 32));
    float mn = fmaxf(mx, tmax);
    float alpha = __expf(mx - mn);
    mx = mn;
    float rs = 0.f;
    #pragma unroll
    for(int st=0; st<2; st++){
      #pragma unroll
      for(int v=0; v<8; v++){
        float pv = __expf(s[st][v] - mn);
        rs += pv;
        Pb[wv][r][st*16 + v + hb*8] = bfc(pv);   // P^T row=query, col=key
      }
    }
    rs += __shfl_xor(rs, 16, 32);
    lsum = lsum*alpha + rs;
    #pragma unroll
    for(int c=0;c<4;c++){
      #pragma unroll
      for(int v=0;v<8;v++) o[c][v] *= alpha;     // per-query alpha = per-lane
    }
    // cross-lane trip through LDS (same wave, in-order DS pipe)
    asm volatile("s_wait_dscnt 0" ::: "memory");
    v16bf bp = ld8x2(&Pb[wv][r][kb], &Pb[wv][r][kb + 16]); // B frag
    // O^T += V^T x P^T over 4 output-dim chunks
    #pragma unroll
    for(int c=0;c<4;c++){
      const __bf16* p = Vp + (size_t)(c*16 + r)*LL + j0 + kb;
      v16bf av = ld8x2(p, p + 16);     // A frag: lane = dim row
      o[c] = __builtin_amdgcn_wmma_f32_16x16x32_bf16(
          false, av, false, bp, (short)0, o[c], false, false);
    }
  }
  // normalize + store: out[b, l, h*64 + d]
  float inv = 1.f / lsum;
  int row = q0 + r;
  #pragma unroll
  for(int c=0;c<4;c++){
    #pragma unroll
    for(int v=0;v<8;v++){
      int d = c*16 + v + hb*8;
      O[((size_t)(b*LL + row))*DDIM + h*HD + d] = bfc(o[c][v]*inv);
    }
  }
}

// ---------------------------------------------------------------------------
extern "C" void kernel_launch(void* const* d_in, const int* in_sizes, int n_in,
                              void* d_out, int out_size, void* d_ws, size_t ws_size,
                              hipStream_t stream) {
  const float* x        = (const float*)d_in[0];
  const float* cond     = (const float*)d_in[1];
  const float* pos      = (const float*)d_in[2];
  const float* rms_s    = (const float*)d_in[3];
  const float* W_cond   = (const float*)d_in[4];
  const float* W_qkv    = (const float*)d_in[5];
  const float* W_out    = (const float*)d_in[6];
  float* out = (float*)d_out;

  const size_t DDe = (size_t)DDIM*DDIM;   // 1M elems
  const size_t MDe = (size_t)MM*DDIM;     // 4M elems
  char* ws = (char*)d_ws;
  __bf16* Ws   = (__bf16*)ws;
  __bf16* Wsh  = Ws  + DDe;
  __bf16* Wg   = Wsh + DDe;
  __bf16* Wq   = Wg  + DDe;
  __bf16* Wk   = Wq  + DDe;
  __bf16* Wv   = Wk  + DDe;
  __bf16* Wot  = Wv  + DDe;
  float*  gate  = (float*)(Wot + DDe);
  float*  scale = gate  + MDe;
  float*  shift = scale + MDe;
  __bf16* condb = (__bf16*)(shift + MDe);
  __bf16* xmb   = condb + MDe;
  __bf16* qw    = xmb + MDe;
  __bf16* kw    = qw  + MDe;
  __bf16* vw    = kw  + MDe;
  float*  invr  = (float*)(vw + MDe);
  // aliased buffers (lifetimes disjoint):
  __bf16* Qh      = (__bf16*)scale;   // written by k_rope, after scale consumed
  __bf16* Kx      = (__bf16*)shift;
  __bf16* Vt      = condb;            // condb dead after cond GEMMs
  __bf16* attnout = qw;               // qw dead after k_rope

  // 1) pack weights (bf16, de-interleaved, pre-transposed N-major)
  k_packw<<<2048, 256, 0, stream>>>(W_cond, Ws,  DDIM, DDIM, 3, 0);
  k_packw<<<2048, 256, 0, stream>>>(W_cond, Wsh, DDIM, DDIM, 3, 1);
  k_packw<<<2048, 256, 0, stream>>>(W_cond, Wg,  DDIM, DDIM, 3, 2);
  k_packw<<<2048, 256, 0, stream>>>(W_qkv,  Wq,  DDIM, DDIM, 3, 0);
  k_packw<<<2048, 256, 0, stream>>>(W_qkv,  Wk,  DDIM, DDIM, 3, 1);
  k_packw<<<2048, 256, 0, stream>>>(W_qkv,  Wv,  DDIM, DDIM, 3, 2);
  k_packw<<<2048, 256, 0, stream>>>(W_out,  Wot, DDIM, DDIM, 1, 0);

  // 2) cond -> bf16 ; rms stats
  k_cvt_bf16<<<4096, 256, 0, stream>>>(cond, condb, (int)MDe);
  k_rms<<<MM, 256, 0, stream>>>(x, invr);

  // 3) cond GEMMs -> scale, shift, gate (f32)
  dim3 gg(DDIM/64, MM/128);
  k_gemm<EPI_F32><<<gg, 256, 0, stream>>>(condb, Ws,  scale, nullptr, nullptr, nullptr, MM, DDIM, DDIM);
  k_gemm<EPI_F32><<<gg, 256, 0, stream>>>(condb, Wsh, shift, nullptr, nullptr, nullptr, MM, DDIM, DDIM);
  k_gemm<EPI_F32><<<gg, 256, 0, stream>>>(condb, Wg,  gate,  nullptr, nullptr, nullptr, MM, DDIM, DDIM);

  // 4) modulate -> xm (bf16)
  k_mod<<<4096, 256, 0, stream>>>(x, invr, rms_s, scale, shift, xmb, (int)MDe);

  // 5) qkv GEMMs (bf16 out)
  k_gemm<EPI_BF16><<<gg, 256, 0, stream>>>(xmb, Wq, nullptr, qw, nullptr, nullptr, MM, DDIM, DDIM);
  k_gemm<EPI_BF16><<<gg, 256, 0, stream>>>(xmb, Wk, nullptr, kw, nullptr, nullptr, MM, DDIM, DDIM);
  k_gemm<EPI_BF16><<<gg, 256, 0, stream>>>(xmb, Wv, nullptr, vw, nullptr, nullptr, MM, DDIM, DDIM);

  // 6) RoPE + head regroup (Q/K -> [b,h,l,d], V -> [b,h,d,l])
  k_rope<<<MM, 256, 0, stream>>>(qw, kw, vw, pos, Qh, Kx, Vt);

  // 7) flash attention
  dim3 ga(LL/64, BB*HH);
  k_attn<<<ga, 128, 0, stream>>>(Qh, Kx, Vt, attnout);

  // 8) output projection + residual + gate (fused epilogue into d_out)
  k_gemm<EPI_FINAL><<<gg, 256, 0, stream>>>(attnout, Wot, out, nullptr, x, gate, MM, DDIM, DDIM);
}
